// SnG_31662498906136
// MI455X (gfx1250) — matrix operand order repaired and checked by reading, not compiled, not run on hardware
//
#include <hip/hip_runtime.h>
#include <hip/hip_bf16.h>

typedef __attribute__((ext_vector_type(16))) _Float16 v16h;
typedef __attribute__((ext_vector_type(8)))  _Float16 v8h;
typedef __attribute__((ext_vector_type(8)))  float    v8f;

// ---------------------------------------------------------------------------
// WMMA helpers (CDNA5 wave32, V_WMMA_F32_16X16X32_F16)
// Fragment layouts used by these helpers (same contiguous k order in A and B):
//  A (16x32 f16):  lane L holds row m=L&15.  hi = L>=16.
//    elem j -> k = kk + (hi?16:0) + j
//  B (32x16 f16):  lane L holds col n=L&15.  elem j -> k = kk + (hi?16:0) + j
//  C/D (16x16 f32): lane L holds col n=L&15; elem r -> row m = r + (hi?8:0)
// All K dims are multiples of 32 by construction -> no guards in hot loops.
// ---------------------------------------------------------------------------
__device__ __forceinline__ v8f wmma16(v16h a, v16h b, v8f c) {
  return __builtin_amdgcn_wmma_f32_16x16x32_f16(false, a, false, b, (short)0, c,
                                                false, false);
}

// A fragment from row-major f16 [M x K]: two contiguous 16B loads.
__device__ __forceinline__ v16h load_a_f16(const _Float16* __restrict__ A,
                                           int lda, int row, int kk, int hi) {
  const _Float16* p = A + (long long)row * lda + kk + (hi ? 16 : 0);
  v8h lo  = *(const v8h*)p;
  v8h hi8 = *(const v8h*)(p + 8);
  v16h a;
#pragma unroll
  for (int j = 0; j < 8; ++j) { a[j] = lo[j]; a[j + 8] = hi8[j]; }
  return a;
}

// A fragment from row-major f32 [M x K] (guard-free, contiguous 8-float runs).
__device__ __forceinline__ v16h load_a_f32ng(const float* __restrict__ A,
                                             int lda, int row, int kk, int hi) {
  const float* p = A + (long long)row * lda + kk + (hi ? 16 : 0);
  v16h a;
#pragma unroll
  for (int j = 0; j < 8; ++j) {
    a[j]     = (_Float16)p[j];
    a[j + 8] = (_Float16)p[8 + j];
  }
  return a;
}

// B fragment from COLUMN-major f16 Bt[n*Kpad + k]: one contiguous 32B load.
__device__ __forceinline__ v16h load_bt_f16(const _Float16* __restrict__ Bt,
                                            int Kpad, int col, int kk, int hi) {
  const _Float16* p = Bt + (long long)col * Kpad + kk + (hi ? 16 : 0);
  return *(const v16h*)p;
}

// ---------------------------------------------------------------------------
// Generic WMMA GEMM: 16x32 output tile per wave.
//  A: f16 row-major [M x K];  Bt: f16 col-major [N x K]; K % 32 == 0.
//  C32 (nullable) f32 [ldc], C16 (nullable) f16 [ldc], both at column coff.
// ---------------------------------------------------------------------------
__global__ void gemm_wmma_f16(const _Float16* __restrict__ A,
                              const _Float16* __restrict__ Bt,
                              const float* __restrict__ bias,
                              float* __restrict__ C32,
                              _Float16* __restrict__ C16,
                              int M, int N, int K, int ldc, int coff, int relu) {
  int lane = threadIdx.x & 31;
  int wave = (int)((blockIdx.x * (long long)blockDim.x + threadIdx.x) >> 5);
  int ntn = N >> 5;                  // 32-col tiles
  int ntm = M >> 4;
  if (wave >= ntm * ntn) return;
  int tm = wave / ntn, tn = wave % ntn;
  int hi = lane >> 4;
  int arow = tm * 16 + (lane & 15);
  int col0 = tn * 32 + (lane & 15);
  int col1 = col0 + 16;
  v8f acc0 = {}, acc1 = {};
  for (int kk = 0; kk < K; kk += 32) {
    v16h a  = load_a_f16(A, K, arow, kk, hi);
    v16h b0 = load_bt_f16(Bt, K, col0, kk, hi);
    v16h b1 = load_bt_f16(Bt, K, col1, kk, hi);
    acc0 = wmma16(a, b0, acc0);
    acc1 = wmma16(a, b1, acc1);
  }
  float bs0 = bias[col0], bs1 = bias[col1];
#pragma unroll
  for (int r = 0; r < 8; ++r) {
    int rr = tm * 16 + r + (hi ? 8 : 0);
    float v0 = acc0[r] + bs0;
    float v1 = acc1[r] + bs1;
    if (relu) { v0 = fmaxf(v0, 0.f); v1 = fmaxf(v1, 0.f); }
    if (C32) {
      C32[(long long)rr * ldc + coff + col0] = v0;
      C32[(long long)rr * ldc + coff + col1] = v1;
    }
    if (C16) {
      C16[(long long)rr * ldc + coff + col0] = (_Float16)v0;
      C16[(long long)rr * ldc + coff + col1] = (_Float16)v1;
    }
  }
}

// ---------------------------------------------------------------------------
// Drug branch: embedding gather + Conv1d(100ch->32ch,k=8) as WMMA GEMM.
// out[b,o,h] = sum_{i,t} emb[xd[b,i], h+t] * conv_w[o,i,t],  h in [0,121)
// A = conv_w16 [32 x 800] f16 row-major;  B[k,n] = e_lds[k>>3][n + (k&7)].
// Stores conv_out16 f16 [B, 3872] (= reshape(32,121), bias added).
// ---------------------------------------------------------------------------
__global__ void drug_conv_kernel(const int* __restrict__ xd,
                                 const float* __restrict__ emb,
                                 const _Float16* __restrict__ conv_w16,
                                 const float* __restrict__ conv_b,
                                 _Float16* __restrict__ conv_out16) {
  __shared__ __align__(16) _Float16 e_lds[100 * 128];
  int b = blockIdx.x;
  int tid = threadIdx.x;  // 128 threads = 4 waves
  for (int idx = tid; idx < 100 * 128; idx += 128) {
    int i = idx >> 7, h = idx & 127;
    e_lds[idx] = (_Float16)emb[xd[b * 100 + i] * 128 + h];
  }
  __syncthreads();
  int lane = tid & 31, wave = tid >> 5, hi = lane >> 4;
  for (int t = wave; t < 16; t += 4) {         // 2 M-tiles x 8 N-tiles
    int tm = t >> 3, tn = t & 7;
    int mrow = tm * 16 + (lane & 15);          // output channel 0..31
    int ncol = tn * 16 + (lane & 15);          // output position
    bool valid = (ncol <= 120);
    int nc = valid ? ncol : 0;                 // clamp (result discarded)
    v8f acc = {};
    for (int kk = 0; kk < 800; kk += 32) {
      v16h a = load_a_f16(conv_w16, 800, mrow, kk, hi);
      v16h bf;
      int base = kk + (hi ? 16 : 0);
      int i0 = base >> 3;                      // base multiple of 8
#pragma unroll
      for (int j = 0; j < 8; ++j) {
        bf[j]     = e_lds[i0 * 128 + nc + j];
        bf[j + 8] = e_lds[(i0 + 1) * 128 + nc + j];
      }
      acc = wmma16(a, bf, acc);
    }
    if (valid) {
#pragma unroll
      for (int r = 0; r < 8; ++r) {
        int m = tm * 16 + r + (hi ? 8 : 0);
        conv_out16[(long long)b * 3872 + m * 121 + ncol] =
            (_Float16)(acc[r] + conv_b[m]);
      }
    }
  }
}

// ---------------------------------------------------------------------------
// Conversion / init utilities
// ---------------------------------------------------------------------------
__global__ void cvt_f16_kernel(const float* __restrict__ s,
                               _Float16* __restrict__ d, long long n) {
  long long i = blockIdx.x * (long long)blockDim.x + threadIdx.x;
  long long st = (long long)gridDim.x * blockDim.x;
  for (; i < n; i += st) d[i] = (_Float16)s[i];
}

// Bt[n*Kpad + k] = (k < K) ? (f16)B[k*N + n] : 0
__global__ void cvt_transpose_f16_kernel(const float* __restrict__ B,
                                         _Float16* __restrict__ Bt,
                                         int K, int N, int Kpad) {
  long long i = blockIdx.x * (long long)blockDim.x + threadIdx.x;
  if (i >= (long long)N * Kpad) return;
  int n = (int)(i / Kpad), k = (int)(i % Kpad);
  Bt[i] = (k < K) ? (_Float16)B[(long long)k * N + n] : (_Float16)0.f;
}

__global__ void zero_f32_kernel(float* __restrict__ d, long long n) {
  long long i = blockIdx.x * (long long)blockDim.x + threadIdx.x;
  long long st = (long long)gridDim.x * blockDim.x;
  for (; i < n; i += st) d[i] = 0.f;
}

// dst[n*dld + f] = src[n*sld + f], f < cols (stride-remap copy for padding)
__global__ void copy_strided_kernel(const float* __restrict__ s, int sld,
                                    float* __restrict__ d, int dld,
                                    int rows, int cols) {
  long long i = blockIdx.x * (long long)blockDim.x + threadIdx.x;
  if (i >= (long long)rows * cols) return;
  int n = (int)(i / cols), f = (int)(i % cols);
  d[(long long)n * dld + f] = s[(long long)n * sld + f];
}

// ---------------------------------------------------------------------------
// Edge scatter-add: agg[dst] += h[src].  f32 atomics (this stage is the
// random-traffic roofline limiter; h fits in the 192MB L2).
// Vector float4 gather path when strides allow (layers >= 1: F=32).
// ---------------------------------------------------------------------------
__global__ void edge_scatter_kernel(const int* __restrict__ src,
                                    const int* __restrict__ dst,
                                    const float* __restrict__ h, int hld,
                                    float* __restrict__ agg, int ald,
                                    int F, long long E) {
  long long e = blockIdx.x * (long long)blockDim.x + threadIdx.x;
  if (e >= E) return;
  const float* hp = h + (long long)src[e] * hld;
  float* ap = agg + (long long)dst[e] * ald;
  if (((hld | ald | F) & 3) == 0) {
    for (int f = 0; f < F; f += 4) {
      float4 v = *(const float4*)(hp + f);
      atomicAdd(&ap[f + 0], v.x);
      atomicAdd(&ap[f + 1], v.y);
      atomicAdd(&ap[f + 2], v.z);
      atomicAdd(&ap[f + 3], v.w);
    }
  } else {
    for (int f = 0; f < F; ++f) atomicAdd(&ap[f], hp[f]);
  }
}

// global_add_pool: pooled[batch[n], f..f+3] += h[n, f..f+3]
__global__ void pool_kernel(const float* __restrict__ h,
                            const int* __restrict__ batch,
                            float* __restrict__ pooled, int N) {
  long long i = blockIdx.x * (long long)blockDim.x + threadIdx.x;
  if (i >= (long long)N * 8) return;
  int n = (int)(i >> 3), f = (int)((i & 7) << 2);
  float4 v = *(const float4*)(h + (long long)n * 32 + f);
  float* pp = pooled + batch[n] * 32 + f;
  atomicAdd(pp + 0, v.x);
  atomicAdd(pp + 1, v.y);
  atomicAdd(pp + 2, v.z);
  atomicAdd(pp + 3, v.w);
}

// ---------------------------------------------------------------------------
// Fused GIN MLP + ReLU + eval-BatchNorm. 16 rows per wave.
// agg: f32 [N x Fpad] (Fpad = 64 or 32, zero padded).  w1t/w2t: f16 col-major.
// out: f32 [N x 32].  agg_next (nullable): f32 [N x 32] pre-init for the next
// layer's aggregation (each wave touches only its own rows: reads of agg are
// consumed by WMMA before the epilogue stores, so in-place per-row is safe).
// ---------------------------------------------------------------------------
__global__ void gin_mlp_kernel(const float* __restrict__ agg, int Fpad,
                               const _Float16* __restrict__ w1t,
                               const float* __restrict__ b1,
                               const _Float16* __restrict__ w2t,
                               const float* __restrict__ b2,
                               const float* __restrict__ bng,
                               const float* __restrict__ bnb,
                               const float* __restrict__ bnm,
                               const float* __restrict__ bnv,
                               float* __restrict__ out,
                               float* __restrict__ agg_next, int Nrows) {
  __shared__ __align__(16) _Float16 hstage[4][16 * 32];  // per-wave staging
  int lane = threadIdx.x & 31;
  int wl = threadIdx.x >> 5;
  int wave = (int)((blockIdx.x * (long long)blockDim.x + threadIdx.x) >> 5);
  int r0 = wave * 16;
  if (r0 >= Nrows) return;
  int hi = lane >> 4;
  int c0 = lane & 15, c1 = 16 + (lane & 15);
  int mrow = r0 + (lane & 15);

  // GEMM1: [16,Fpad] x [Fpad,32]
  v8f acc0 = {}, acc1 = {};
  for (int kk = 0; kk < Fpad; kk += 32) {
    v16h a   = load_a_f32ng(agg, Fpad, mrow, kk, hi);
    v16h b0  = load_bt_f16(w1t, Fpad, c0, kk, hi);
    v16h b1f = load_bt_f16(w1t, Fpad, c1, kk, hi);
    acc0 = wmma16(a, b0, acc0);
    acc1 = wmma16(a, b1f, acc1);
  }
  // bias + relu, stage to LDS (wave-private; same wave consumes -> in order)
  _Float16* hb = hstage[wl];
  float bias0 = b1[c0], bias1 = b1[c1];
#pragma unroll
  for (int r = 0; r < 8; ++r) {
    int rr = r + (hi ? 8 : 0);
    hb[rr * 32 + c0] = (_Float16)fmaxf(acc0[r] + bias0, 0.f);
    hb[rr * 32 + c1] = (_Float16)fmaxf(acc1[r] + bias1, 0.f);
  }
  v16h a2;
  {
    int m = lane & 15, base = hi ? 16 : 0;
    v8h lo  = *(const v8h*)&hb[m * 32 + base];
    v8h hi8 = *(const v8h*)&hb[m * 32 + 8 + base];
#pragma unroll
    for (int j = 0; j < 8; ++j) { a2[j] = lo[j]; a2[j + 8] = hi8[j]; }
  }
  v16h b20 = load_bt_f16(w2t, 32, c0, 0, hi);
  v16h b21 = load_bt_f16(w2t, 32, c1, 0, hi);
  v8f z = {};
  v8f o0 = wmma16(a2, b20, z);
  v8f o1 = wmma16(a2, b21, z);

  // epilogue: +b2, relu, BN(eval)
  float s0 = rsqrtf(bnv[c0] + 1e-5f), s1 = rsqrtf(bnv[c1] + 1e-5f);
  float g0 = bng[c0] * s0, g1 = bng[c1] * s1;
  float m0 = bnm[c0], m1 = bnm[c1];
  float a0 = bnb[c0], a1 = bnb[c1];
  float bb20 = b2[c0], bb21 = b2[c1];
#pragma unroll
  for (int r = 0; r < 8; ++r) {
    int rr = r0 + r + (hi ? 8 : 0);
    float v0 = fmaxf(o0[r] + bb20, 0.f);
    float v1 = fmaxf(o1[r] + bb21, 0.f);
    v0 = g0 * (v0 - m0) + a0;
    v1 = g1 * (v1 - m1) + a1;
    long long i0 = (long long)rr * 32 + c0;
    long long i1 = (long long)rr * 32 + c1;
    out[i0] = v0;
    out[i1] = v1;
    if (agg_next) { agg_next[i0] = v0; agg_next[i1] = v1; }
  }
}

// ---------------------------------------------------------------------------
// Final 256->1 projection + emit (out, y) tuple.
// ---------------------------------------------------------------------------
__global__ void fc3_kernel(const float* __restrict__ h2,
                           const float* __restrict__ w,
                           const float* __restrict__ b,
                           const float* __restrict__ y,
                           float* __restrict__ out) {
  int i = blockIdx.x * blockDim.x + threadIdx.x;
  if (i >= 256) return;
  float s = b[0];
  const float* row = h2 + i * 256;
  for (int k = 0; k < 256; ++k) s += row[k] * w[k];
  out[i] = s;
  out[256 + i] = y[i];
}

// ---------------------------------------------------------------------------
extern "C" void kernel_launch(void* const* d_in, const int* in_sizes, int n_in,
                              void* d_out, int out_size, void* d_ws, size_t ws_size,
                              hipStream_t stream) {
  const int*   xd     = (const int*)d_in[0];
  const float* xt     = (const float*)d_in[1];
  const int*   edge   = (const int*)d_in[2];
  const int*   batch  = (const int*)d_in[3];
  const float* y      = (const float*)d_in[4];
  const float* emb    = (const float*)d_in[5];
  const float* conv_w = (const float*)d_in[6];
  const float* conv_b = (const float*)d_in[7];
  const float* fcd_w  = (const float*)d_in[8];
  const float* fcd_b  = (const float*)d_in[9];
  const float* g1_w1  = (const float*)d_in[10];
  const float* g1_b1  = (const float*)d_in[11];
  const float* g1_w2  = (const float*)d_in[12];
  const float* g1_b2  = (const float*)d_in[13];
  const float* gw1    = (const float*)d_in[14];
  const float* gb1    = (const float*)d_in[15];
  const float* gw2    = (const float*)d_in[16];
  const float* gb2    = (const float*)d_in[17];
  const float* bn_g   = (const float*)d_in[18];
  const float* bn_b   = (const float*)d_in[19];
  const float* bn_rm  = (const float*)d_in[20];
  const float* bn_rv  = (const float*)d_in[21];
  const float* fct_w  = (const float*)d_in[22];
  const float* fct_b  = (const float*)d_in[23];
  const float* c1_w   = (const float*)d_in[24];
  const float* c1_b   = (const float*)d_in[25];
  const float* c2_w   = (const float*)d_in[26];
  const float* c2_b   = (const float*)d_in[27];
  const float* c3_w   = (const float*)d_in[28];
  const float* c3_b   = (const float*)d_in[29];

  const long long E = in_sizes[2] / 2;
  const int N = in_sizes[3];     // 100000

  // workspace layout (offsets 64KB-aligned)
  char* ws = (char*)d_ws;
  const size_t MB = 1024 * 1024;
  _Float16* conv_out16 = (_Float16*)(ws);              // 256*3872 f16 (~2MB)
  _Float16* xj16       = (_Float16*)(ws + 2 * MB);     // 256*256 f16
  _Float16* h1_16      = (_Float16*)(ws + 3 * MB);     // 256*1024 f16
  float*    h2         = (float*)   (ws + 4 * MB);     // 256*256 f32
  float*    pooled     = (float*)   (ws + 5 * MB);     // 256*32 f32
  _Float16* pooled16   = (_Float16*)(ws + 5 * MB + 64 * 1024);
  _Float16* fcd_wt     = (_Float16*)(ws + 6 * MB);     // 128 x 3872 (~1MB)
  _Float16* c1_wt      = (_Float16*)(ws + 7 * MB);     // 1024 x 256
  _Float16* c2_wt      = (_Float16*)(ws + 8 * MB);     // 256 x 1024
  _Float16* fct_wt     = (_Float16*)(ws + 9 * MB);     // 128 x 32
  _Float16* conv_w16   = (_Float16*)(ws + 9 * MB + 128 * 1024); // 32*800
  _Float16* ginw       = (_Float16*)(ws + 9 * MB + 512 * 1024); // 5x(w1t,w2t)
  float*    aggb64     = (float*)   (ws + 12 * MB);    // N*64 f32 (~25.6MB)
  float*    aggb32     = (float*)   (ws + 38 * MB);    // N*32 f32 (~12.8MB)
  float*    hA         = (float*)   (ws + 51 * MB);    // N*32 f32
  float*    hB         = (float*)   (ws + 64 * MB);    // N*32 f32

  auto cvt = [&](const float* s, _Float16* d, long long n) {
    cvt_f16_kernel<<<(int)((n + 255) / 256), 256, 0, stream>>>(s, d, n);
  };
  auto cvtT = [&](const float* B, _Float16* Bt, int K, int Nn, int Kpad) {
    long long n = (long long)Nn * Kpad;
    cvt_transpose_f16_kernel<<<(int)((n + 255) / 256), 256, 0, stream>>>(B, Bt, K, Nn, Kpad);
  };
  auto gemm = [&](const _Float16* A, const _Float16* Bt, const float* bias,
                  float* C32, _Float16* C16, int M, int Nn, int K,
                  int ldc, int coff, int relu) {
    int tiles = (M >> 4) * (Nn >> 5);
    gemm_wmma_f16<<<(tiles + 3) / 4, 128, 0, stream>>>(A, Bt, bias, C32, C16,
                                                       M, Nn, K, ldc, coff, relu);
  };

  // ---- weight pre-conversion (f16, B transposed to col-major) ----
  cvt(conv_w, conv_w16, 32 * 800);
  cvtT(fcd_w, fcd_wt, 3872, 128, 3872);
  cvtT(fct_w, fct_wt, 32, 128, 32);
  cvtT(c1_w, c1_wt, 256, 1024, 256);
  cvtT(c2_w, c2_wt, 1024, 256, 1024);
  cvtT(g1_w1, ginw, 41, 32, 64);                    // layer0 w1t (padded K)
  cvtT(g1_w2, ginw + 2048, 32, 32, 32);             // layer0 w2t
  for (int l = 1; l < 5; ++l) {
    cvtT(gw1 + (l - 1) * 1024, ginw + l * 4096, 32, 32, 32);
    cvtT(gw2 + (l - 1) * 1024, ginw + l * 4096 + 2048, 32, 32, 32);
  }

  // ---- drug branch ----
  drug_conv_kernel<<<256, 128, 0, stream>>>(xd, emb, conv_w16, conv_b, conv_out16);
  gemm(conv_out16, fcd_wt, fcd_b, nullptr, xj16, 256, 128, 3872, 256, 0, 0);

  // ---- protein branch: 5 GIN layers ----
  int mlp_blocks = ((N >> 4) + 3) / 4;              // 6250 waves
  int sc_blocks = (int)((E + 255) / 256);

  // layer 0: padded [N,64] agg from xt
  zero_f32_kernel<<<2048, 256, 0, stream>>>(aggb64, (long long)N * 64);
  {
    long long nc = (long long)N * 41;
    copy_strided_kernel<<<(int)((nc + 255) / 256), 256, 0, stream>>>(
        xt, 41, aggb64, 64, N, 41);
  }
  edge_scatter_kernel<<<sc_blocks, 256, 0, stream>>>(
      edge, edge + E, xt, 41, aggb64, 64, 41, E);
  gin_mlp_kernel<<<mlp_blocks, 128, 0, stream>>>(
      aggb64, 64, ginw, g1_b1, ginw + 2048, g1_b2,
      bn_g, bn_b, bn_rm, bn_rv, hA, aggb32, N);     // h -> hA, and pre-init agg

  // layers 1..4: agg pre-initialized by previous MLP epilogue
  const float* hin = hA;
  float* cur = hB;
  for (int l = 1; l < 5; ++l) {
    edge_scatter_kernel<<<sc_blocks, 256, 0, stream>>>(
        edge, edge + E, hin, 32, aggb32, 32, 32, E);
    float* agg_next = (l < 4) ? aggb32 : nullptr;   // in-place per-row (safe)
    gin_mlp_kernel<<<mlp_blocks, 128, 0, stream>>>(
        aggb32, 32, ginw + l * 4096, gb1 + (l - 1) * 32,
        ginw + l * 4096 + 2048, gb2 + (l - 1) * 32,
        bn_g + l * 32, bn_b + l * 32, bn_rm + l * 32, bn_rv + l * 32,
        cur, agg_next, N);
    const float* t = cur; cur = (float*)hin; hin = t;
  }

  // ---- pool + protein FC ----
  zero_f32_kernel<<<32, 256, 0, stream>>>(pooled, 256 * 32);
  pool_kernel<<<(int)(((long long)N * 8 + 255) / 256), 256, 0, stream>>>(
      hin, batch, pooled, N);
  cvt(pooled, pooled16, 256 * 32);
  gemm(pooled16, fct_wt, fct_b, nullptr, xj16, 256, 128, 32, 256, 128, 1);

  // ---- classifier ----
  gemm(xj16, c1_wt, c1_b, nullptr, h1_16, 256, 1024, 256, 1024, 0, 1);
  gemm(h1_16, c2_wt, c2_b, h2, nullptr, 256, 256, 1024, 256, 0, 1);
  fc3_kernel<<<1, 256, 0, stream>>>(h2, c3_w, c3_b, y, (float*)d_out);
}
